// GATLayer_47201690583090
// MI455X (gfx1250) — matrix operand order repaired and checked by reading, not compiled, not run on hardware
//
#include <hip/hip_runtime.h>
#include <hip/hip_bf16.h>
#include <math.h>

typedef __attribute__((ext_vector_type(16))) __bf16 v16bf;
typedef __attribute__((ext_vector_type(8)))  float  v8f;

#define IN_DIM 256
#define HC     256
#define NHEAD  4
#define NEG_SLOPE 0.2f
#define BN_EPS 1e-5f

// monotonic float<->uint key for atomic max on floats (handles negatives)
__device__ __forceinline__ unsigned f2key(float f) {
  unsigned b = __float_as_uint(f);
  return (b & 0x80000000u) ? ~b : (b | 0x80000000u);
}
__device__ __forceinline__ float key2f(unsigned k) {
  return (k & 0x80000000u) ? __uint_as_float(k ^ 0x80000000u) : __uint_as_float(~k);
}

// force native global_atomic_add_f32 (no CAS expansion)
__device__ __forceinline__ void atomAddF(float* p, float v) { unsafeAtomicAdd(p, v); }

__device__ __forceinline__ float lrelu(float e) { return (e > 0.f) ? e : NEG_SLOPE * e; }

// ---------------------------------------------------------------------------
// Kernel 0: pack W [256x256 f32, row-major (k, hc)] into WMMA B-operand bf16
// layout: WB[ct][ks][lane][e]  with  B[k][n] : k = ks*32 + (e<8 ? 8*(lane>>4) + e : 16 + 8*(lane>>4) + e-8),
// n = ct*16 + (lane&15)   (CDNA5 ISA B layout, wave32)
// ---------------------------------------------------------------------------
__global__ void pack_W(const float* __restrict__ W, __bf16* __restrict__ WB) {
  int t = blockIdx.x * blockDim.x + threadIdx.x;   // 16*8*32 = 4096 packers
  if (t >= 16 * 8 * 32) return;
  int lane = t & 31;
  int ks   = (t >> 5) & 7;
  int ct   = t >> 8;
  int half = lane >> 4, n = lane & 15;
  __bf16* dst = WB + (size_t)t * 16;
#pragma unroll
  for (int e = 0; e < 16; ++e) {
    int k = ks * 32 + ((e < 8) ? 8 * half + e : 16 + 8 * half + (e - 8));
    dst[e] = (__bf16)W[(size_t)k * HC + ct * 16 + n];
  }
}

// ---------------------------------------------------------------------------
// Kernel 1: x = feature @ W via v_wmma_f32_16x16x32_bf16.
// One wave per 16x16 output tile; 8 K-steps of 32.
// ---------------------------------------------------------------------------
__global__ void __launch_bounds__(256) gemm_x(const float* __restrict__ feature,
                                              const __bf16* __restrict__ WB,
                                              float* __restrict__ x,
                                              int row_tiles) {
  int wid  = threadIdx.x >> 5;
  int lane = threadIdx.x & 31;
  int tile = blockIdx.x * 8 + wid;
  if (tile >= row_tiles * 16) return;
  int rt = tile >> 4, ct = tile & 15;
  int half = lane >> 4, m = lane & 15;

  const float*  arow = feature + (size_t)(rt * 16 + m) * IN_DIM;
  const v16bf*  wbv  = (const v16bf*)WB;

  v8f acc = {};
#pragma unroll
  for (int ks = 0; ks < 8; ++ks) {
    // A operand: 16-bit A 16x32 layout. VGPR v holds K pairs:
    // v<4 : K = 2v,2v+1 (+8 for lanes16-31) ; v>=4 : K = 16+2(v-4).. (+8)
    v16bf a;
#pragma unroll
    for (int v = 0; v < 8; ++v) {
      int base = (v < 4) ? 0 : 16;
      int k = ks * 32 + base + 8 * half + 2 * (v & 3);
      float2 f = *(const float2*)(arow + k);
      a[2 * v]     = (__bf16)f.x;
      a[2 * v + 1] = (__bf16)f.y;
    }
    v16bf b = wbv[(size_t)(ct * 8 + ks) * 32 + lane];
    acc = __builtin_amdgcn_wmma_f32_16x16x32_bf16(
        /*neg_a=*/false, a, /*neg_b=*/false, b,
        /*c_mod=*/(short)0, acc, /*reuse_a=*/false, /*reuse_b=*/false);
  }
  // D layout: reg r, lane -> row = r + 8*half, col = lane&15
#pragma unroll
  for (int r = 0; r < 8; ++r)
    x[(size_t)(rt * 16 + r + 8 * half) * HC + ct * 16 + m] = acc[r];
}

// ---------------------------------------------------------------------------
// Kernel 2: per-node attention logits a_src/a_dst [N, H]
// ---------------------------------------------------------------------------
__global__ void attn_logits(const float* __restrict__ x,
                            const float* __restrict__ att_src,
                            const float* __restrict__ att_dst,
                            float* __restrict__ a_src, float* __restrict__ a_dst) {
  __shared__ float ss[256], sd[256];
  int node = blockIdx.x, t = threadIdx.x;
  float v = x[(size_t)node * HC + t];
  ss[t] = v * att_src[t];
  sd[t] = v * att_dst[t];
  __syncthreads();
  for (int off = 32; off >= 1; off >>= 1) {
    if ((t & 63) < off) { ss[t] += ss[t + off]; sd[t] += sd[t + off]; }
    __syncthreads();
  }
  if ((t & 63) == 0) {
    int h = t >> 6;
    a_src[node * NHEAD + h] = ss[t];
    a_dst[node * NHEAD + h] = sd[t];
  }
}

// init: mkey = key(-inf), denom = 0
__global__ void init_softmax(unsigned* __restrict__ mkey, float* __restrict__ denom, int cnt) {
  int i = blockIdx.x * blockDim.x + threadIdx.x;
  if (i < cnt) { mkey[i] = 0x007FFFFFu; denom[i] = 0.f; }
}

// ---------------------------------------------------------------------------
// Kernel 3a/3b: per-dst segment max, then segment sum of exp.
// Per-node head quads are 16B-contiguous -> single b128 loads.
// ---------------------------------------------------------------------------
__global__ void edge_max(const int* __restrict__ ei, int E, int n,
                         const float* __restrict__ a_src, const float* __restrict__ a_dst,
                         unsigned* __restrict__ mkey) {
  int i = blockIdx.x * blockDim.x + threadIdx.x;
  int T = E + n;
  if (i >= T) return;
  int s = (i < E) ? ei[i]     : (i - E);
  int d = (i < E) ? ei[E + i] : (i - E);
  float4 as = *(const float4*)(a_src + (size_t)s * NHEAD);
  float4 ad = *(const float4*)(a_dst + (size_t)d * NHEAD);
  unsigned* mk = mkey + (size_t)d * NHEAD;
  atomicMax(mk + 0, f2key(lrelu(as.x + ad.x)));
  atomicMax(mk + 1, f2key(lrelu(as.y + ad.y)));
  atomicMax(mk + 2, f2key(lrelu(as.z + ad.z)));
  atomicMax(mk + 3, f2key(lrelu(as.w + ad.w)));
}

__global__ void edge_denom(const int* __restrict__ ei, int E, int n,
                           const float* __restrict__ a_src, const float* __restrict__ a_dst,
                           const unsigned* __restrict__ mkey, float* __restrict__ denom) {
  int i = blockIdx.x * blockDim.x + threadIdx.x;
  int T = E + n;
  if (i >= T) return;
  int s = (i < E) ? ei[i]     : (i - E);
  int d = (i < E) ? ei[E + i] : (i - E);
  float4 as = *(const float4*)(a_src + (size_t)s * NHEAD);
  float4 ad = *(const float4*)(a_dst + (size_t)d * NHEAD);
  uint4  mk = *(const uint4*)(mkey + (size_t)d * NHEAD);
  float* dn = denom + (size_t)d * NHEAD;
  atomAddF(dn + 0, expf(lrelu(as.x + ad.x) - key2f(mk.x)));
  atomAddF(dn + 1, expf(lrelu(as.y + ad.y) - key2f(mk.y)));
  atomAddF(dn + 2, expf(lrelu(as.z + ad.z) - key2f(mk.z)));
  atomAddF(dn + 3, expf(lrelu(as.w + ad.w) - key2f(mk.w)));
}

// ---------------------------------------------------------------------------
// Kernel 3c: weighted aggregation. One wave per edge; lane owns 8 channels.
// x fits in L2 (51MB < 192MB) -> gather is L2-resident.
// Lanes 0..3 compute the 4 per-head alphas; broadcast via wave32 shuffle.
// ---------------------------------------------------------------------------
__global__ void __launch_bounds__(256) edge_aggregate(
    const int* __restrict__ ei, int E, int n,
    const float* __restrict__ a_src, const float* __restrict__ a_dst,
    const unsigned* __restrict__ mkey, const float* __restrict__ denom,
    const float* __restrict__ x, float* __restrict__ outagg) {
  int wid = threadIdx.x >> 5, lane = threadIdx.x & 31;
  int i = blockIdx.x * 8 + wid;
  int T = E + n;
  if (i >= T) return;
  int s = (i < E) ? ei[i]     : (i - E);
  int d = (i < E) ? ei[E + i] : (i - E);

  const float4* xr = (const float4*)(x + (size_t)s * HC) + lane * 2;
  __builtin_prefetch(xr, 0, 0);   // global_prefetch_b8

  float av = 0.f;
  if (lane < NHEAD) {             // one lane per head computes alpha
    float e = lrelu(a_src[(size_t)s * NHEAD + lane] + a_dst[(size_t)d * NHEAD + lane]);
    av = expf(e - key2f(mkey[(size_t)d * NHEAD + lane])) / denom[(size_t)d * NHEAD + lane];
  }
  int h = lane >> 3;              // 8 lanes per head (64 ch / 8 per lane)
  float alpha = __shfl(av, h, 32);

  float4 x0 = xr[0], x1 = xr[1];
  float* o = outagg + (size_t)d * HC + lane * 8;
  atomAddF(o + 0, x0.x * alpha); atomAddF(o + 1, x0.y * alpha);
  atomAddF(o + 2, x0.z * alpha); atomAddF(o + 3, x0.w * alpha);
  atomAddF(o + 4, x1.x * alpha); atomAddF(o + 5, x1.y * alpha);
  atomAddF(o + 6, x1.z * alpha); atomAddF(o + 7, x1.w * alpha);
}

// ---------------------------------------------------------------------------
// Kernel 4: batchnorm statistics (thread t = channel t, block = 256-row strip)
// ---------------------------------------------------------------------------
__global__ void bn_stats(const float* __restrict__ outagg, const float* __restrict__ bias,
                         float* __restrict__ sums, float* __restrict__ sumsq, int n) {
  int t = threadIdx.x;
  int r0 = blockIdx.x * 256;
  int r1 = r0 + 256; if (r1 > n) r1 = n;
  float b = bias[t], s = 0.f, q = 0.f;
  for (int r = r0; r < r1; ++r) {
    float v = outagg[(size_t)r * HC + t] + b;
    s += v; q += v * v;
  }
  atomAddF(&sums[t], s);
  atomAddF(&sumsq[t], q);
}

// ---------------------------------------------------------------------------
// Kernel 5: normalize + ELU + residual -> d_out
// ---------------------------------------------------------------------------
__global__ void finalize(const float* __restrict__ outagg, const float* __restrict__ feature,
                         const float* __restrict__ bias, const float* __restrict__ gamma,
                         const float* __restrict__ beta, const float* __restrict__ sums,
                         const float* __restrict__ sumsq, float* __restrict__ out, int n) {
  int node = blockIdx.x, c = threadIdx.x;
  float invN = 1.0f / (float)n;
  float mu  = sums[c] * invN;
  float var = sumsq[c] * invN - mu * mu;
  float v  = outagg[(size_t)node * HC + c] + bias[c];
  float hn = (v - mu) * rsqrtf(var + BN_EPS) * gamma[c] + beta[c];
  float el = (hn > 0.f) ? hn : expm1f(hn);
  out[(size_t)node * HC + c] = feature[(size_t)node * HC + c] + el;
}

// ---------------------------------------------------------------------------
extern "C" void kernel_launch(void* const* d_in, const int* in_sizes, int n_in,
                              void* d_out, int out_size, void* d_ws, size_t ws_size,
                              hipStream_t stream) {
  const float* feature = (const float*)d_in[0];
  const int*   ei      = (const int*)d_in[1];   // [2, E] (JAX x64 off -> int32)
  const float* W       = (const float*)d_in[2];
  const float* att_src = (const float*)d_in[3];
  const float* att_dst = (const float*)d_in[4];
  const float* bias    = (const float*)d_in[5];
  const float* gamma   = (const float*)d_in[6];
  const float* beta    = (const float*)d_in[7];
  float* out = (float*)d_out;

  int n = in_sizes[0] / IN_DIM;   // 50000
  int E = in_sizes[1] / 2;        // 800000

  // workspace carve-up (256B aligned)
  char* ws = (char*)d_ws;
  size_t off = 0;
  auto carve = [&](size_t bytes) -> void* {
    void* p = ws + off;
    off = (off + bytes + 255) & ~(size_t)255;
    return p;
  };
  float*    x      = (float*)   carve((size_t)n * HC * sizeof(float));
  __bf16*   WB     = (__bf16*)  carve((size_t)16 * 8 * 32 * 16 * sizeof(__bf16));
  float*    a_src  = (float*)   carve((size_t)n * NHEAD * sizeof(float));
  float*    a_dst  = (float*)   carve((size_t)n * NHEAD * sizeof(float));
  unsigned* mkey   = (unsigned*)carve((size_t)n * NHEAD * sizeof(unsigned));
  float*    denom  = (float*)   carve((size_t)n * NHEAD * sizeof(float));
  float*    sums   = (float*)   carve(HC * sizeof(float));
  float*    sumsq  = (float*)   carve(HC * sizeof(float));
  float*    outagg = (float*)   carve((size_t)n * HC * sizeof(float));

  // per-launch re-init (graph-replay safe)
  hipMemsetAsync(outagg, 0, (size_t)n * HC * sizeof(float), stream);
  hipMemsetAsync(sums,   0, HC * sizeof(float), stream);
  hipMemsetAsync(sumsq,  0, HC * sizeof(float), stream);
  init_softmax<<<(n * NHEAD + 255) / 256, 256, 0, stream>>>(mkey, denom, n * NHEAD);

  // 0) pack W into WMMA B layout
  pack_W<<<16, 256, 0, stream>>>(W, WB);

  // 1) x = feature @ W  (WMMA bf16)
  int row_tiles = n / 16;                         // 50000 % 16 == 0
  int tiles = row_tiles * 16;
  gemm_x<<<(tiles + 7) / 8, 256, 0, stream>>>(feature, WB, x, row_tiles);

  // 2) attention logits
  attn_logits<<<n, 256, 0, stream>>>(x, att_src, att_dst, a_src, a_dst);

  // 3) segment softmax + aggregation over E + N self-loop edges
  int T = E + n;
  edge_max  <<<(T + 255) / 256, 256, 0, stream>>>(ei, E, n, a_src, a_dst, mkey);
  edge_denom<<<(T + 255) / 256, 256, 0, stream>>>(ei, E, n, a_src, a_dst, mkey, denom);
  edge_aggregate<<<(T + 7) / 8, 256, 0, stream>>>(ei, E, n, a_src, a_dst, mkey, denom, x, outagg);

  // 4) batchnorm stats, 5) normalize + ELU + residual
  bn_stats<<<(n + 255) / 256, 256, 0, stream>>>(outagg, bias, sums, sumsq, n);
  finalize<<<n, 256, 0, stream>>>(outagg, feature, bias, gamma, beta, sums, sumsq, out, n);
}
